// SerialBlock_adapt_37142877176326
// MI455X (gfx1250) — matrix operand order repaired and checked by reading, not compiled, not run on hardware
//
#include <hip/hip_runtime.h>
#include <hip/hip_bf16.h>

typedef __attribute__((ext_vector_type(16))) _Float16 v16h;
typedef __attribute__((ext_vector_type(8)))  float    v8f;

union H16 { uint4 u[2]; v16h h; };

#define SCALEc 0.17677669529663689f   // 32^-0.5

#define WMMA_F16(A_, B_, C_) \
  __builtin_amdgcn_wmma_f32_16x16x32_f16(false, (A_), false, (B_), (short)0, (C_), false, false)

// ---------------------------------------------------------------------------
// f32 -> f16 convert
// ---------------------------------------------------------------------------
__global__ __launch_bounds__(256) void k_cvt(const float* __restrict__ s,
                                             _Float16* __restrict__ d, int n) {
  int i = blockIdx.x * 256 + threadIdx.x;
  if (i < n) d[i] = (_Float16)s[i];
}

// ---------------------------------------------------------------------------
// GEMM1: qkv = xh[32768,256] @ qkv_w[768,256]^T + bias -> f16 [32768,768]
// 2x2 register-blocked: one wave per 32x32 output tile (4 wmma / k-step).
// 8 waves of a block share tn (B tile hot in WGP$), consecutive tm.
// ---------------------------------------------------------------------------
__global__ __launch_bounds__(256) void k_gemm_qkv(
    const _Float16* __restrict__ A,   // [32768,256]
    const _Float16* __restrict__ Bw,  // [768,256] row-major (out,in)
    const float*    __restrict__ bias,
    _Float16*       __restrict__ out) // [32768,768]
{
  const int M = 32768, N = 768, K = 256;
  int wave = threadIdx.x >> 5;
  int lane = threadIdx.x & 31;
  int tid  = blockIdx.x * 8 + wave;
  const int tilesM = M / 32;          // 1024
  int tm = (tid % tilesM) * 32;
  int tn = (tid / tilesM) * 32;
  int hs = lane >> 4;                 // half-wave select
  int lr = lane & 15;
  const _Float16* arow0 = A  + (size_t)(tm      + lr) * K;
  const _Float16* arow1 = A  + (size_t)(tm + 16 + lr) * K;
  const _Float16* brow0 = Bw + (size_t)(tn      + lr) * K;
  const _Float16* brow1 = Bw + (size_t)(tn + 16 + lr) * K;
  v8f acc00 = {}, acc01 = {}, acc10 = {}, acc11 = {};
#pragma unroll
  for (int k = 0; k < K; k += 32) {
    int a0 = k + hs * 8;
    int b0 = k + hs * 8;
    H16 af0, af1, bf0, bf1;
    af0.u[0] = *(const uint4*)(arow0 + a0);
    af0.u[1] = *(const uint4*)(arow0 + a0 + 16);
    af1.u[0] = *(const uint4*)(arow1 + a0);
    af1.u[1] = *(const uint4*)(arow1 + a0 + 16);
    bf0.u[0] = *(const uint4*)(brow0 + b0);
    bf0.u[1] = *(const uint4*)(brow0 + b0 + 16);
    bf1.u[0] = *(const uint4*)(brow1 + b0);
    bf1.u[1] = *(const uint4*)(brow1 + b0 + 16);
    __builtin_prefetch(arow0 + a0 + 32, 0, 3);
    acc00 = WMMA_F16(af0.h, bf0.h, acc00);
    acc01 = WMMA_F16(af0.h, bf1.h, acc01);
    acc10 = WMMA_F16(af1.h, bf0.h, acc10);
    acc11 = WMMA_F16(af1.h, bf1.h, acc11);
  }
  int   n0 = tn + lr, n1 = tn + 16 + lr;
  float bb0 = bias[n0], bb1 = bias[n1];
#pragma unroll
  for (int r = 0; r < 8; ++r) {
    int m0 = tm + hs * 8 + r;
    int m1 = tm + 16 + hs * 8 + r;
    out[(size_t)m0 * N + n0] = (_Float16)(acc00[r] + bb0);
    out[(size_t)m0 * N + n1] = (_Float16)(acc01[r] + bb1);
    out[(size_t)m1 * N + n0] = (_Float16)(acc10[r] + bb0);
    out[(size_t)m1 * N + n1] = (_Float16)(acc11[r] + bb1);
  }
}

// ---------------------------------------------------------------------------
// Column softmax of k over N (per b,h,ch) + ktv = k_softmax^T v  [B,8,32,32]
// one block per (b,h)
// ---------------------------------------------------------------------------
__global__ __launch_bounds__(256) void k_softmax_ktv(
    const _Float16* __restrict__ qkvh,  // [32768,768] (q|k|v slabs)
    float* __restrict__ ktv)            // [8,8,32,32]
{
  int bh = blockIdx.x;
  int b = bh >> 3, h = bh & 7;
  int t = threadIdx.x;
  int col = t & 31, seg = t >> 5;
  __shared__ float sm[256], ss[256];
  __shared__ float colM[32], colInv[32];
  const _Float16* kbase = qkvh + (size_t)b * 4096 * 768 + 256 + h * 32;
  const _Float16* vbase = qkvh + (size_t)b * 4096 * 768 + 512 + h * 32;

  // pass 1: streaming max / sum-of-exp per column
  float m = -3.0e38f, s = 0.f;
  for (int n = seg * 512; n < seg * 512 + 512; ++n) {
    float x  = (float)kbase[(size_t)n * 768 + col];
    float nm = fmaxf(m, x);
    s = s * __expf(m - nm) + __expf(x - nm);
    m = nm;
  }
  sm[t] = m; ss[t] = s;
  __syncthreads();
  if (t < 32) {
    float M = sm[t];
    for (int g = 1; g < 8; ++g) M = fmaxf(M, sm[g * 32 + t]);
    float S = 0.f;
    for (int g = 0; g < 8; ++g) S += ss[g * 32 + t] * __expf(sm[g * 32 + t] - M);
    colM[t] = M;
    colInv[t] = 1.f / S;
  }
  __syncthreads();

  // pass 2: accumulate ktv in 16-row LDS chunks
  __shared__ float ek[16 * 32], vv[16 * 32];
  float acc[4] = {0.f, 0.f, 0.f, 0.f};
  for (int n0 = 0; n0 < 4096; n0 += 16) {
#pragma unroll
    for (int i = 0; i < 2; ++i) {
      int j = t + i * 256;
      int r = j >> 5, cc = j & 31;
      size_t off = (size_t)(n0 + r) * 768 + cc;
      ek[j] = __expf((float)kbase[off] - colM[cc]) * colInv[cc];
      vv[j] = (float)vbase[off];
    }
    __syncthreads();
#pragma unroll
    for (int i = 0; i < 4; ++i) {
      int idx = t + i * 256;
      int chk = idx >> 5, chv = idx & 31;
#pragma unroll
      for (int r = 0; r < 16; ++r) acc[i] += ek[r * 32 + chk] * vv[r * 32 + chv];
    }
    __syncthreads();
  }
#pragma unroll
  for (int i = 0; i < 4; ++i) {
    int idx = t + i * 256;
    int chk = idx >> 5, chv = idx & 31;
    ktv[(size_t)(((b * 8 + h) * 32 + chk) * 32 + chv)] = acc[i];
  }
}

// ---------------------------------------------------------------------------
// depthwise conv templates for CRPE (3/5/7)
// ---------------------------------------------------------------------------
template <int KS>
__device__ __forceinline__ void crpe_conv(const _Float16* lv, int c,
                                          const float* wp, float bias,
                                          float* outv) {
  float w[KS * KS];
#pragma unroll
  for (int i = 0; i < KS * KS; ++i) w[i] = wp[i];
  const int R = KS / 2;
#pragma unroll
  for (int xi = 0; xi < 8; ++xi) {
    float s = bias;
#pragma unroll
    for (int dy = -R; dy <= R; ++dy)
#pragma unroll
      for (int dx = -R; dx <= R; ++dx)
        s += w[(dy + R) * KS + (dx + R)] *
             (float)lv[((3 + dy) * 14 + (3 + xi + dx)) * 256 + c];
    outv[xi] = s;
  }
}

// ---------------------------------------------------------------------------
// factor_att = SCALE * q@ktv + q * dwconv(v)   -> fa f32 [B,N,256]
// block = (b, y, x0..x0+7); LDS tile of v (7x14x256) + q row (8x256)
// ---------------------------------------------------------------------------
__global__ __launch_bounds__(256) void k_crpe_fa(
    const _Float16* __restrict__ qkvh, const float* __restrict__ ktv,
    const float* __restrict__ w3, const float* __restrict__ b3,
    const float* __restrict__ w5, const float* __restrict__ b5,
    const float* __restrict__ w7, const float* __restrict__ b7,
    float* __restrict__ fa)
{
  __shared__ unsigned int lv_u32[98 * 128];  // 50176 B: v halo tile (f16)
  __shared__ unsigned int lq_u32[8 * 128];   //  4096 B: q row tile (f16)
  const _Float16* lv = (const _Float16*)lv_u32;
  const _Float16* lq = (const _Float16*)lq_u32;
  int t = threadIdx.x;
  int bix = blockIdx.x;
  int x0 = (bix & 7) * 8;
  int y  = (bix >> 3) & 63;
  int b  = bix >> 9;

  for (int idx = t; idx < 98 * 128; idx += 256) {
    int p = idx >> 7, pr = idx & 127;
    int ry = p / 14, rx = p % 14;
    int gy = y + ry - 3, gx = x0 + rx - 3;
    unsigned int val = 0u;
    if (gy >= 0 && gy < 64 && gx >= 0 && gx < 64) {
      size_t row = (size_t)(b * 4096 + gy * 64 + gx);
      val = *(const unsigned int*)(qkvh + row * 768 + 512 + pr * 2);
    }
    lv_u32[idx] = val;
  }
  for (int idx = t; idx < 8 * 128; idx += 256) {
    int p = idx >> 7, pr = idx & 127;
    size_t row = (size_t)(b * 4096 + y * 64 + x0 + p);
    lq_u32[idx] = *(const unsigned int*)(qkvh + row * 768 + pr * 2);
  }
  __syncthreads();

  int c = t;
  int h = c >> 5, ch = c & 31;
  float kc[32];
#pragma unroll
  for (int j = 0; j < 32; ++j)
    kc[j] = ktv[(size_t)(((b * 8 + h) * 32 + j) * 32 + ch)];

  float conv[8];
  if (c < 64)       crpe_conv<3>(lv, c, w3 + c * 9,          b3[c],       conv);
  else if (c < 160) crpe_conv<5>(lv, c, w5 + (c - 64) * 25,  b5[c - 64],  conv);
  else              crpe_conv<7>(lv, c, w7 + (c - 160) * 49, b7[c - 160], conv);

#pragma unroll
  for (int xi = 0; xi < 8; ++xi) {
    const _Float16* qrow = lq + xi * 256;
    float dot = 0.f;
#pragma unroll
    for (int j = 0; j < 32; ++j) dot += (float)qrow[(h << 5) + j] * kc[j];
    float qv = (float)qrow[c];
    fa[(size_t)(b * 4096 + y * 64 + x0 + xi) * 256 + c] = SCALEc * dot + qv * conv[xi];
  }
}

// ---------------------------------------------------------------------------
// pooled partial sums over N (deterministic, no atomics)
// ---------------------------------------------------------------------------
__global__ __launch_bounds__(256) void k_pool(const float* __restrict__ fa,
                                              float* __restrict__ part) {
  int bix = blockIdx.x;
  int b = bix >> 3, c8 = bix & 7;
  int t = threadIdx.x;
  float s = 0.f;
  for (int n = c8 * 512; n < c8 * 512 + 512; ++n)
    s += fa[(size_t)(b * 4096 + n) * 256 + t];
  part[(size_t)bix * 256 + t] = s;
}

// ---------------------------------------------------------------------------
// SE MLP: pooled -> relu(pooled@w1^T) -> sigmoid(hid@w2^T); gf = 1+gate
// ---------------------------------------------------------------------------
__global__ __launch_bounds__(256) void k_se(const float* __restrict__ part,
                                            const float* __restrict__ w1,
                                            const float* __restrict__ w2,
                                            float* __restrict__ gf) {
  __shared__ float pooled[8 * 256];
  __shared__ float hid[8 * 128];
  int t = threadIdx.x;
  for (int idx = t; idx < 2048; idx += 256) {
    int b = idx >> 8, c = idx & 255;
    float s = 0.f;
    for (int g = 0; g < 8; ++g) s += part[(size_t)(b * 8 + g) * 256 + c];
    pooled[idx] = s * (1.0f / 4096.0f);
  }
  __syncthreads();
  for (int idx = t; idx < 1024; idx += 256) {
    int b = idx >> 7, j = idx & 127;
    float d = 0.f;
    for (int c = 0; c < 256; ++c) d += pooled[b * 256 + c] * w1[j * 256 + c];
    hid[idx] = fmaxf(d, 0.f);
  }
  __syncthreads();
  if (t < 64) {
    int b = t >> 3, hh = t & 7;
    float d = 0.f;
    for (int j = 0; j < 128; ++j) d += hid[b * 128 + j] * w2[hh * 128 + j];
    gf[t] = 1.f + 1.f / (1.f + __expf(-d));
  }
}

// ---------------------------------------------------------------------------
// GEMM3: out = (fa * (1+gate)) @ proj_w^T + proj_b -> f32 d_out
// 2x2 register-blocked, gate fused into A-fragment load (f32 -> f16 on fly)
// ---------------------------------------------------------------------------
__device__ __forceinline__ v16h gated_frag(const float* __restrict__ arow,
                                           int a0, float g) {
  float4 f0 = *(const float4*)(arow + a0);
  float4 f1 = *(const float4*)(arow + a0 + 4);
  float4 f2 = *(const float4*)(arow + a0 + 16);
  float4 f3 = *(const float4*)(arow + a0 + 20);
  v16h av;
  av[0]  = (_Float16)(f0.x * g); av[1]  = (_Float16)(f0.y * g);
  av[2]  = (_Float16)(f0.z * g); av[3]  = (_Float16)(f0.w * g);
  av[4]  = (_Float16)(f1.x * g); av[5]  = (_Float16)(f1.y * g);
  av[6]  = (_Float16)(f1.z * g); av[7]  = (_Float16)(f1.w * g);
  av[8]  = (_Float16)(f2.x * g); av[9]  = (_Float16)(f2.y * g);
  av[10] = (_Float16)(f2.z * g); av[11] = (_Float16)(f2.w * g);
  av[12] = (_Float16)(f3.x * g); av[13] = (_Float16)(f3.y * g);
  av[14] = (_Float16)(f3.z * g); av[15] = (_Float16)(f3.w * g);
  return av;
}

__global__ __launch_bounds__(256) void k_gemm_proj(
    const float*    __restrict__ A,    // fa [32768,256]
    const float*    __restrict__ gf,   // [64] (1+gate)
    const _Float16* __restrict__ Bw,   // projh [256,256]
    const float*    __restrict__ bias, // [256]
    float*          __restrict__ out)  // [32768,256]
{
  const int M = 32768, N = 256, K = 256;
  int wave = threadIdx.x >> 5;
  int lane = threadIdx.x & 31;
  int tid  = blockIdx.x * 8 + wave;
  const int tilesM = M / 32;          // 1024
  int tm = (tid % tilesM) * 32;
  int tn = (tid / tilesM) * 32;
  int hs = lane >> 4;
  int lr = lane & 15;
  const float*    arow0 = A  + (size_t)(tm      + lr) * K;
  const float*    arow1 = A  + (size_t)(tm + 16 + lr) * K;
  const _Float16* brow0 = Bw + (size_t)(tn      + lr) * K;
  const _Float16* brow1 = Bw + (size_t)(tn + 16 + lr) * K;
  int bidx = tm >> 12;                // batch index, constant within tile
  float gr[8];
#pragma unroll
  for (int j = 0; j < 8; ++j) gr[j] = gf[bidx * 8 + j];
  v8f acc00 = {}, acc01 = {}, acc10 = {}, acc11 = {};
#pragma unroll
  for (int k = 0; k < K; k += 32) {
    float g = gr[k >> 5];             // both 8-wide chunks lie in head k/32
    int a0 = k + hs * 8;
    int b0 = k + hs * 8;
    v16h af0 = gated_frag(arow0, a0, g);
    v16h af1 = gated_frag(arow1, a0, g);
    H16 bf0, bf1;
    bf0.u[0] = *(const uint4*)(brow0 + b0);
    bf0.u[1] = *(const uint4*)(brow0 + b0 + 16);
    bf1.u[0] = *(const uint4*)(brow1 + b0);
    bf1.u[1] = *(const uint4*)(brow1 + b0 + 16);
    __builtin_prefetch(arow0 + a0 + 32, 0, 3);
    acc00 = WMMA_F16(af0, bf0.h, acc00);
    acc01 = WMMA_F16(af0, bf1.h, acc01);
    acc10 = WMMA_F16(af1, bf0.h, acc10);
    acc11 = WMMA_F16(af1, bf1.h, acc11);
  }
  int   n0 = tn + lr, n1 = tn + 16 + lr;
  float bb0 = bias[n0], bb1 = bias[n1];
#pragma unroll
  for (int r = 0; r < 8; ++r) {
    int m0 = tm + hs * 8 + r;
    int m1 = tm + 16 + hs * 8 + r;
    out[(size_t)m0 * N + n0] = acc00[r] + bb0;
    out[(size_t)m0 * N + n1] = acc01[r] + bb1;
    out[(size_t)m1 * N + n0] = acc10[r] + bb0;
    out[(size_t)m1 * N + n1] = acc11[r] + bb1;
  }
}

// ---------------------------------------------------------------------------
extern "C" void kernel_launch(void* const* d_in, const int* in_sizes, int n_in,
                              void* d_out, int out_size, void* d_ws, size_t ws_size,
                              hipStream_t stream) {
  (void)in_sizes; (void)n_in; (void)out_size; (void)ws_size;
  const float* x      = (const float*)d_in[0];
  const float* qkv_w  = (const float*)d_in[1];
  const float* qkv_b  = (const float*)d_in[2];
  const float* proj_w = (const float*)d_in[3];
  const float* proj_b = (const float*)d_in[4];
  const float* se_w1  = (const float*)d_in[5];
  const float* se_w2  = (const float*)d_in[6];
  const float* w3     = (const float*)d_in[7];
  const float* b3     = (const float*)d_in[8];
  const float* w5     = (const float*)d_in[9];
  const float* b5     = (const float*)d_in[10];
  const float* w7     = (const float*)d_in[11];
  const float* b7     = (const float*)d_in[12];

  char* w = (char*)d_ws;
  size_t off = 0;
  _Float16* xh     = (_Float16*)(w + off); off += 32768ull * 256 * 2;   // 16.8 MB
  _Float16* wqkvh  = (_Float16*)(w + off); off += 768ull * 256 * 2;
  _Float16* wprojh = (_Float16*)(w + off); off += 256ull * 256 * 2;
  _Float16* qkvh   = (_Float16*)(w + off); off += 32768ull * 768 * 2;   // 50.3 MB
  float*    fa     = (float*)(w + off);    off += 32768ull * 256 * 4;   // 33.6 MB
  float*    ktv    = (float*)(w + off);    off += 8ull * 8 * 32 * 32 * 4;
  float*    part   = (float*)(w + off);    off += 64ull * 256 * 4;
  float*    gf     = (float*)(w + off);    off += 64ull * 4;

  k_cvt<<<(8388608 + 255) / 256, 256, 0, stream>>>(x, xh, 8388608);
  k_cvt<<<(196608  + 255) / 256, 256, 0, stream>>>(qkv_w, wqkvh, 196608);
  k_cvt<<<(65536   + 255) / 256, 256, 0, stream>>>(proj_w, wprojh, 65536);

  // 1024 x 24 (32x32) tiles, 8 waves/block
  k_gemm_qkv<<<3072, 256, 0, stream>>>(xh, wqkvh, qkv_b, qkvh);
  k_softmax_ktv<<<64, 256, 0, stream>>>(qkvh, ktv);
  k_crpe_fa<<<4096, 256, 0, stream>>>(qkvh, ktv, w3, b3, w5, b5, w7, b7, fa);
  k_pool<<<64, 256, 0, stream>>>(fa, part);
  k_se<<<1, 256, 0, stream>>>(part, se_w1, se_w2, gf);
  // 1024 x 8 (32x32) tiles, 8 waves/block
  k_gemm_proj<<<1024, 256, 0, stream>>>(fa, gf, wprojh, proj_b, (float*)d_out);
}
